// DreamGraphReasoner_55405078119324
// MI455X (gfx1250) — matrix operand order repaired and checked by reading, not compiled, not run on hardware
//
#include <hip/hip_runtime.h>

typedef __bf16 bf16;
typedef __attribute__((ext_vector_type(16))) __bf16 v16bf;
typedef __attribute__((ext_vector_type(8)))  __bf16 v8bf;
typedef __attribute__((ext_vector_type(8)))  float  v8f;

#define D_DIM   512
#define N_NODES 1024
#define B_DIM   16
#define M_ROWS  (B_DIM * N_NODES)   // 16384
#define G_DREAMS 16
#define L_STEPS  64
#define H_HOPS   3

// ---------------------------------------------------------------------------
// Prep: nodes = (what+action+result)/3, layout (G,L,B,D) -> x[b][node][d]
// ---------------------------------------------------------------------------
__global__ __launch_bounds__(256)
void prep_nodes(const float* __restrict__ what, const float* __restrict__ action,
                const float* __restrict__ result, float* __restrict__ x,
                bf16* __restrict__ xb)
{
    int t = blockIdx.x * 256 + threadIdx.x;        // 0 .. 8388607
    int d    = t & 511;
    int b    = (t >> 9) & 15;
    int node = t >> 13;                            // g*64 + l
    size_t src = ((size_t)node * 16 + b) * 512 + d;
    float v = (what[src] + action[src] + result[src]) * (1.0f / 3.0f);
    size_t dst = ((size_t)b * N_NODES + node) * 512 + d;
    x[dst]  = v;
    xb[dst] = (bf16)v;
}

// ---------------------------------------------------------------------------
// Prep: transpose+convert the six 512x512 f32 weights to bf16 Wt[n][k]=W[k][n]
// z = 0..2 -> Wq/Wk/Wv ; z = 3..5 -> W_hop[0..2]
// ---------------------------------------------------------------------------
__global__ __launch_bounds__(256)
void prep_weights(const float* __restrict__ Wq, const float* __restrict__ Wk,
                  const float* __restrict__ Wv, const float* __restrict__ Whop,
                  bf16* __restrict__ wt_qkv, bf16* __restrict__ wt_hop)
{
    int t = blockIdx.x * 256 + threadIdx.x;        // 0 .. 512*512-1
    int z = blockIdx.y;
    int k = t >> 9, n = t & 511;
    const float* src = (z == 0) ? Wq : (z == 1) ? Wk : (z == 2) ? Wv
                                 : (Whop + (size_t)(z - 3) * D_DIM * D_DIM);
    bf16* dst = (z < 3) ? (wt_qkv + (size_t)z * D_DIM * D_DIM)
                        : (wt_hop + (size_t)(z - 3) * D_DIM * D_DIM);
    dst[(size_t)n * D_DIM + k] = (bf16)src[(size_t)k * D_DIM + n];
}

// ---------------------------------------------------------------------------
// WMMA GEMM: C[M,512] = A[M,512](bf16, row major) @ Wt^T (Wt stored [n][k] bf16)
// The block's 64x512 weight tile (64 KB) is staged into LDS ONCE via
// CDNA5 async copies (global_load_async_to_lds_b128 / s_wait_asynccnt),
// then all 8 waves feed their B fragments from LDS (ds_load_b128).
// MODE 0 (QKV): out[z][m][n] = acc + bias_z[n]        (grid.z selects Q/K/V)
// MODE 1 (hop): x[m][n] += relu(acc + bias[n]); refresh x_bf16
// Wave tile 16x64 (4 accumulators), block = 8 waves -> 128x64, K step 32.
// ---------------------------------------------------------------------------
template <int MODE>
__global__ __launch_bounds__(256)
void gemm_wmma(const bf16* __restrict__ A, const bf16* __restrict__ WtBase,
               const float* __restrict__ b0, const float* __restrict__ b1,
               const float* __restrict__ b2, float* __restrict__ outBase,
               const float* __restrict__ xin, float* __restrict__ xout,
               bf16* __restrict__ xoutb)
{
    __shared__ bf16 ldsB[64 * D_DIM];            // 64 KB weight tile

    const int tid  = threadIdx.x;
    const int lane = tid & 31;
    const int wv   = tid >> 5;
    const int half = lane >> 4;      // 0 or 1
    const int lcol = lane & 15;
    const int z    = blockIdx.z;

    const bf16*  Wt   = WtBase + (size_t)z * D_DIM * D_DIM;
    const float* bias = (MODE == 0) ? ((z == 0) ? b0 : (z == 1) ? b1 : b2) : b0;
    float* out = (MODE == 0) ? (outBase + (size_t)z * (size_t)M_ROWS * D_DIM) : nullptr;

    const int m0 = blockIdx.x * 128 + wv * 16;
    const int n0 = blockIdx.y * 64;

    // ---- async stage of the 64x512 bf16 weight tile into LDS (64 KB) ----
    {
        const unsigned lbase = (unsigned)(uintptr_t)(&ldsB[0]);   // LDS byte offset
        const unsigned long long gbase =
            (unsigned long long)(uintptr_t)(Wt + (size_t)n0 * D_DIM);
#pragma unroll
        for (int i = 0; i < 16; ++i) {
            const unsigned off = ((unsigned)tid + 256u * (unsigned)i) * 16u; // 16 B chunks
            const unsigned laddr = lbase + off;
            const unsigned long long gaddr = gbase + off;
            asm volatile("global_load_async_to_lds_b128 %0, %1, off"
                         :: "v"(laddr), "v"(gaddr) : "memory");
        }
        asm volatile("s_wait_asynccnt 0x0" ::: "memory");
    }
    __syncthreads();

    v8f acc[4] = {};
    const int arow = m0 + lcol;      // A row held by this lane (ISA A layout)

    for (int k0 = 0; k0 < D_DIM; k0 += 32) {
        // A fragment: row 'arow', K = kb..kb+15, kb = half*16
        union { v16bf v; v8bf h[2]; } a;
        const bf16* ap = A + (size_t)arow * D_DIM + k0 + half * 16;
        a.h[0] = *(const v8bf*)(ap);
        a.h[1] = *(const v8bf*)(ap + 8);
#pragma unroll
        for (int t = 0; t < 4; ++t) {
            // B fragment from LDS: local col t*16+lcol, K = k0 + half*16 .. +16
            const bf16* bp = &ldsB[(size_t)(t * 16 + lcol) * D_DIM + k0 + half * 16];
            v16bf bfrag = *(const v16bf*)(bp);   // 32 B, bank-conflict-free ds reads
            acc[t] = __builtin_amdgcn_wmma_f32_16x16x32_bf16(
                false, a.v, false, bfrag, (short)0, acc[t], false, false);
        }
    }

#pragma unroll
    for (int t = 0; t < 4; ++t) {
        const int col = n0 + t * 16 + lcol;
        const float bb = bias[col];
#pragma unroll
        for (int r = 0; r < 8; ++r) {
            const int row = m0 + half * 8 + r;   // ISA C/D layout
            const size_t idx = (size_t)row * D_DIM + col;
            float v = acc[t][r] + bb;
            if (MODE == 0) {
                out[idx] = v;
            } else {
                float rel = v > 0.0f ? v : 0.0f;
                float nx  = xin[idx] + rel;
                xout[idx]  = nx;
                xoutb[idx] = (bf16)nx;
            }
        }
    }
}

// ---------------------------------------------------------------------------
// Sparse masked attention: one wave32 per (b,node) row. 15 causal neighbors
// (other dreams, same step) + 1 temporal (same dream, step+1, if any).
// Lane owns a 16-element D-chunk; dot products reduced via __shfl_xor.
// ---------------------------------------------------------------------------
__global__ __launch_bounds__(256)
void attn_sparse(const float* __restrict__ Q, const float* __restrict__ K,
                 const float* __restrict__ V, bf16* __restrict__ attOut)
{
    const int lane = threadIdx.x & 31;
    const int wv   = threadIdx.x >> 5;
    const int r    = blockIdx.x * 8 + wv;    // row in [0, 16384)
    const int b     = r >> 10;
    const int node  = r & 1023;
    const int dream = node >> 6;
    const int s     = node & 63;
    const int bbase = b << 10;

    int nb[16];
#pragma unroll
    for (int i = 0; i < 15; ++i) {
        int dd = i + (i >= dream ? 1 : 0);   // skip own dream
        nb[i] = bbase + (dd << 6) + s;
    }
    const bool hasT = (s < L_STEPS - 1);
    nb[15] = hasT ? (bbase + node + 1) : nb[0];

    const int dchunk = lane * 16;
    float q[16];
    {
        const float* qp = Q + (size_t)r * D_DIM + dchunk;
#pragma unroll
        for (int i = 0; i < 16; i += 4) {
            float4 t4 = *(const float4*)(qp + i);
            q[i] = t4.x; q[i + 1] = t4.y; q[i + 2] = t4.z; q[i + 3] = t4.w;
        }
    }

    const float scale = 0.044194173824159216f;   // 1/sqrt(512)
    float sc[16];
#pragma unroll
    for (int j = 0; j < 16; ++j) {
        const float* kp = K + (size_t)nb[j] * D_DIM + dchunk;
        float p = 0.0f;
#pragma unroll
        for (int i = 0; i < 16; i += 4) {
            float4 kv = *(const float4*)(kp + i);
            p += q[i] * kv.x + q[i + 1] * kv.y + q[i + 2] * kv.z + q[i + 3] * kv.w;
        }
#pragma unroll
        for (int off = 16; off > 0; off >>= 1) p += __shfl_xor(p, off, 32);
        sc[j] = p * scale;
    }
    if (!hasT) sc[15] = -1e30f;                  // masked slot -> weight 0

    float mx = sc[0];
#pragma unroll
    for (int j = 1; j < 16; ++j) mx = fmaxf(mx, sc[j]);
    float sum = 0.0f;
#pragma unroll
    for (int j = 0; j < 16; ++j) { sc[j] = __expf(sc[j] - mx); sum += sc[j]; }
    const float inv = 1.0f / sum;

    float accv[16];
#pragma unroll
    for (int i = 0; i < 16; ++i) accv[i] = 0.0f;
#pragma unroll
    for (int j = 0; j < 16; ++j) {
        const float w = sc[j] * inv;
        const float* vp = V + (size_t)nb[j] * D_DIM + dchunk;
#pragma unroll
        for (int i = 0; i < 16; i += 4) {
            float4 vv = *(const float4*)(vp + i);
            accv[i] += w * vv.x; accv[i + 1] += w * vv.y;
            accv[i + 2] += w * vv.z; accv[i + 3] += w * vv.w;
        }
    }
    bf16* op = attOut + (size_t)r * D_DIM + dchunk;
#pragma unroll
    for (int i = 0; i < 16; ++i) op[i] = (bf16)accv[i];
}

// ---------------------------------------------------------------------------
// Head: mean over nodes, then MLP (tiny, exact f32 VALU)
// ---------------------------------------------------------------------------
__global__ __launch_bounds__(256)
void reduce_mean(const float* __restrict__ x, float* __restrict__ agg)
{
    int t = blockIdx.x * 256 + threadIdx.x;      // 0 .. B*D-1
    int b = t >> 9, d = t & 511;
    const float* p = x + (size_t)b * N_NODES * D_DIM + d;
    float s = 0.0f;
    for (int n = 0; n < N_NODES; ++n) s += p[(size_t)n * D_DIM];
    agg[t] = s * (1.0f / (float)N_NODES);
}

__global__ __launch_bounds__(256)
void mlp1(const float* __restrict__ agg, const float* __restrict__ W1,
          const float* __restrict__ bias1, float* __restrict__ hdn)
{
    int t = blockIdx.x * 256 + threadIdx.x;      // 0 .. B*1024-1
    int b = t >> 10, h = t & 1023;
    const float* a = agg + (size_t)b * D_DIM;
    float s = bias1[h];
    for (int d = 0; d < D_DIM; ++d) s += a[d] * W1[(size_t)d * 1024 + h];
    hdn[t] = s > 0.0f ? s : 0.0f;
}

__global__ __launch_bounds__(256)
void mlp2(const float* __restrict__ hdn, const float* __restrict__ W2,
          const float* __restrict__ bias2, float* __restrict__ out)
{
    int t = blockIdx.x * 256 + threadIdx.x;      // 0 .. B*512-1
    int b = t >> 9, d = t & 511;
    const float* h = hdn + (size_t)b * 1024;
    float s = bias2[d];
    for (int k = 0; k < 1024; ++k) s += h[k] * W2[(size_t)k * D_DIM + d];
    out[t] = s;
}

// ---------------------------------------------------------------------------
extern "C" void kernel_launch(void* const* d_in, const int* in_sizes, int n_in,
                              void* d_out, int out_size, void* d_ws, size_t ws_size,
                              hipStream_t stream)
{
    const float* what   = (const float*)d_in[0];
    const float* action = (const float*)d_in[1];
    const float* result = (const float*)d_in[2];
    const float* Wq     = (const float*)d_in[3];
    const float* bq     = (const float*)d_in[4];
    const float* Wk     = (const float*)d_in[5];
    const float* bk     = (const float*)d_in[6];
    const float* Wv     = (const float*)d_in[7];
    const float* bv     = (const float*)d_in[8];
    const float* W_hop  = (const float*)d_in[9];
    const float* b_hop  = (const float*)d_in[10];
    const float* W_agg1 = (const float*)d_in[11];
    const float* b_agg1 = (const float*)d_in[12];
    const float* W_agg2 = (const float*)d_in[13];
    const float* b_agg2 = (const float*)d_in[14];
    float* out = (float*)d_out;

    const size_t MN = (size_t)M_ROWS * D_DIM;    // 8,388,608 elems
    char* ws = (char*)d_ws;
    float* x     = (float*)(ws);                         // 33,554,432 B
    bf16*  xb    = (bf16*)(ws + 33554432ull);            // 16,777,216 B
    float* qkv   = (float*)(ws + 50331648ull);           // 100,663,296 B (Q|K|V)
    bf16*  attb  = (bf16*)(ws + 150994944ull);           // 16,777,216 B
    bf16*  wtqkv = (bf16*)(ws + 167772160ull);           //  1,572,864 B
    bf16*  wthop = (bf16*)(ws + 169345024ull);           //  1,572,864 B
    float* agg   = (float*)(ws + 170917888ull);          //     32,768 B
    float* hdn   = (float*)(ws + 170950656ull);          //     65,536 B

    float* Qm = qkv;
    float* Km = qkv + MN;
    float* Vm = qkv + 2 * MN;

    prep_nodes<<<(M_ROWS * D_DIM) / 256, 256, 0, stream>>>(what, action, result, x, xb);
    prep_weights<<<dim3((D_DIM * D_DIM) / 256, 6), 256, 0, stream>>>(
        Wq, Wk, Wv, W_hop, wtqkv, wthop);

    for (int h = 0; h < H_HOPS; ++h) {
        // QKV projections: 3 WMMA GEMMs via grid.z
        gemm_wmma<0><<<dim3(M_ROWS / 128, D_DIM / 64, 3), 256, 0, stream>>>(
            xb, wtqkv, bq, bk, bv, qkv, nullptr, nullptr, nullptr);
        // sparse masked attention (<=16 neighbors per row)
        attn_sparse<<<M_ROWS / 8, 256, 0, stream>>>(Qm, Km, Vm, attb);
        // hop transform + ReLU + residual (in place on x / x_bf16)
        gemm_wmma<1><<<dim3(M_ROWS / 128, D_DIM / 64, 1), 256, 0, stream>>>(
            attb, wthop + (size_t)h * D_DIM * D_DIM,
            b_hop + (size_t)h * D_DIM, nullptr, nullptr, nullptr, x, x, xb);
    }

    reduce_mean<<<(B_DIM * D_DIM) / 256, 256, 0, stream>>>(x, agg);
    mlp1<<<(B_DIM * 1024) / 256, 256, 0, stream>>>(agg, W_agg1, b_agg1, hdn);
    mlp2<<<(B_DIM * D_DIM) / 256, 256, 0, stream>>>(hdn, W_agg2, b_agg2, out);
}
